// MultiTaskClassificationModel_39642548142227
// MI455X (gfx1250) — matrix operand order repaired and checked by reading, not compiled, not run on hardware
//
#include <hip/hip_runtime.h>
#include <math.h>

typedef __attribute__((ext_vector_type(16))) __bf16 v16bf;
typedef __attribute__((ext_vector_type(8)))  __bf16 v8bf;
typedef __attribute__((ext_vector_type(8)))  float  v8f;
typedef unsigned int u32x4 __attribute__((ext_vector_type(4)));
typedef int          i32x4 __attribute__((ext_vector_type(4)));
typedef int          i32x8 __attribute__((ext_vector_type(8)));

// ---- problem sizes ----
constexpr int kNA  = 100000;
constexpr int kNS  = 32;
constexpr int kNG  = 18;
constexpr int kNEM = 9;
constexpr int kE   = 100000;
constexpr int kD   = 128;   // feature dim of node embeddings / H
constexpr int kB   = 4096;  // batch rows of x
constexpr int kNSH = 3840;  // HH/2 (shared layer width)
constexpr int kCV  = 7552;  // 59*128 flattened task embeddings

// output layout (floats): style | genre | emotion | z_a
constexpr int kOutStyle   = 0;
constexpr int kOutGenre   = kB * kNS;                 // 131072
constexpr int kOutEmotion = kOutGenre + kB * kNG;     // 204800
constexpr int kOutZA      = kOutEmotion + kB * kNEM;  // 241664

// ---- workspace offsets (floats) ----
constexpr int WS_VSRC  = 0;        // 6*128: Wsrc[i]@att_src[i]
constexpr int WS_VDST  = 768;      // 6*128: Wdst[i]@att_dst[i]
constexpr int WS_BSUM  = 1536;     // 128: bias_gat[1]+[3]+[5]
constexpr int WS_HSREV = 1664;     // 59*128: small-node src projections (rev types)
constexpr int WS_DSCAL = 9216;     // 59 (pad 64): fwd dst attention scalars
constexpr int WS_SREVS = 9280;     // 59 (pad 64): rev src attention scalars
constexpr int WS_SFWD  = 9344;     // 3*NA: artwork fwd src scalars
constexpr int WS_SREV  = 309344;   // 3*NA: artwork rev dst scalars
constexpr int WS_DENF  = 609344;   // 64: fwd softmax denominators
constexpr int WS_DENR  = 609408;   // 3*NA: rev softmax denominators
constexpr int WS_WBF   = 909408;   // 3*E: fwd exp(e)
constexpr int WS_WBR   = 1209408;  // 3*E: rev exp(e)
constexpr int WS_ZRAW  = 1509408;  // 59*128: fwd raw aggregation (D-space)
constexpr int WS_CVB   = 1516960;  // 7552: cv = [z_s|z_g|z_e]
constexpr int WS_PART  = 1524512;  // 32*3840: GEMV partials
constexpr int WS_BIASN = 1647392;  // 3840: bh + cv@Wh[128:]

// ---- CDNA5 Tensor Data Mover availability ----
#if defined(__has_builtin)
#if __has_builtin(__builtin_amdgcn_tensor_load_to_lds) && \
    __has_builtin(__builtin_amdgcn_s_wait_tensorcnt)
#define USE_TDM 1
#endif
#endif
#ifndef USE_TDM
#define USE_TDM 0
#endif

__device__ inline v16bf cat16(v8bf lo, v8bf hi) {
  return __builtin_shufflevector(lo, hi, 0,1,2,3,4,5,6,7,8,9,10,11,12,13,14,15);
}

#if USE_TDM
// Issue one TDM descriptor: copy a 2D fp32 tile [rows x cols] (row stride
// `stride` elements) from global `gsrc` into contiguous LDS at `lds_byte`.
// D# layout per CDNA5 ISA 8.3/8.4 (group0: count|lds|gaddr|type=2,
// group1: data_size=4B, tensor dims, tile dims, dim0 stride).
__device__ inline void tdm_load_tile_f32(const float* gsrc, unsigned lds_byte,
                                         unsigned rows, unsigned cols, unsigned stride) {
  unsigned long long ga = (unsigned long long)(uintptr_t)gsrc;
  u32x4 g0;
  g0[0] = 1u;                                             // count=1 (user mode)
  g0[1] = lds_byte;                                       // lds_addr
  g0[2] = (unsigned)(ga & 0xFFFFFFFFu);                   // global_addr[31:0]
  g0[3] = (unsigned)((ga >> 32) & 0x01FFFFFFu) | (2u << 30); // addr[56:32] | type=2
  i32x8 g1;
  g1[0] = (int)(2u << 16);                                // data_size=4B; mask=0
  g1[1] = (int)((stride * 4u & 0xFFFFu) << 16);           // tensor_dim0 lo (use row len)
  g1[2] = (int)(((stride * 4u) >> 16) | ((rows & 0xFFFFu) << 16)); // dim0 hi | dim1 lo
  g1[3] = (int)((rows >> 16) | ((cols & 0xFFFFu) << 16)); // dim1 hi | tile_dim0
  g1[4] = (int)(rows & 0xFFFFu);                          // tile_dim1 (tile_dim2=0)
  g1[5] = (int)stride;                                    // tensor_dim0_stride lo32
  g1[6] = 0;                                              // stride hi | dim1_stride lo
  g1[7] = 0;
  i32x4 gz = {0, 0, 0, 0};
#if defined(__clang_major__) && (__clang_major__ >= 23)
  i32x8 gz8 = {0, 0, 0, 0, 0, 0, 0, 0};
  __builtin_amdgcn_tensor_load_to_lds(g0, g1, gz, gz, gz8, 0);
#else
  __builtin_amdgcn_tensor_load_to_lds(g0, g1, gz, gz, 0);
#endif
}
#endif

// ------------------------------------------------------------------
__global__ void k_zero(float* __restrict__ p, int n) {
  int i = blockIdx.x * blockDim.x + threadIdx.x;
  if (i < n) p[i] = 0.f;
}

// ------------------------------------------------------------------
// Stage A: tiny precompute (one block).
__global__ __launch_bounds__(256) void k_small(
    const float* __restrict__ xs_s, const float* __restrict__ xs_g, const float* __restrict__ xs_e,
    const float* __restrict__ Wsrc, const float* __restrict__ Wdst,
    const float* __restrict__ att_src, const float* __restrict__ att_dst,
    const float* __restrict__ bias_gat, float* __restrict__ ws) {
  float* vsrc  = ws + WS_VSRC;
  float* vdst  = ws + WS_VDST;
  float* bsum  = ws + WS_BSUM;
  float* hsRev = ws + WS_HSREV;
  float* dScal = ws + WS_DSCAL;
  float* sRevS = ws + WS_SREVS;
  const int tid = threadIdx.x;

  for (int idx = tid; idx < 768; idx += 256) {
    int i = idx >> 7, d = idx & 127;
    float a = 0.f, b = 0.f;
    for (int h = 0; h < 128; ++h) {
      a += Wsrc[i * 16384 + d * 128 + h] * att_src[i * 128 + h];
      b += Wdst[i * 16384 + d * 128 + h] * att_dst[i * 128 + h];
    }
    vsrc[idx] = a; vdst[idx] = b;
  }
  for (int h = tid; h < 128; h += 256)
    bsum[h] = bias_gat[128 + h] + bias_gat[384 + h] + bias_gat[640 + h];

  for (int idx = tid; idx < kCV; idx += 256) {   // hsRev = x_small @ Wsrc[1|3|5]
    int row = idx >> 7, h = idx & 127;
    int t, n; const float* xs;
    if (row < 32)      { t = 0; n = row;      xs = xs_s; }
    else if (row < 50) { t = 1; n = row - 32; xs = xs_g; }
    else               { t = 2; n = row - 50; xs = xs_e; }
    int i = 2 * t + 1;
    float a = 0.f;
    for (int d = 0; d < 128; ++d) a += xs[n * 128 + d] * Wsrc[i * 16384 + d * 128 + h];
    hsRev[idx] = a;
  }
  __syncthreads();
  for (int row = tid; row < 59; row += 256) {
    int t, n; const float* xs;
    if (row < 32)      { t = 0; n = row;      xs = xs_s; }
    else if (row < 50) { t = 1; n = row - 32; xs = xs_g; }
    else               { t = 2; n = row - 50; xs = xs_e; }
    float a = 0.f; const float* vd = vdst + (2 * t) * 128;
    for (int d = 0; d < 128; ++d) a += xs[n * 128 + d] * vd[d];
    dScal[row] = a;
    float b = 0.f; const float* as = att_src + (2 * t + 1) * 128;
    for (int h = 0; h < 128; ++h) b += hsRev[row * 128 + h] * as[h];
    sRevS[row] = b;
  }
}

// ------------------------------------------------------------------
// Stage B: one pass over x_artwork, 6 attention scalars per row (wave/row).
__global__ __launch_bounds__(256) void k_rowscal(
    const float* __restrict__ xA, const float* __restrict__ vsrc,
    const float* __restrict__ vdst, float* __restrict__ sFwd, float* __restrict__ sRev) {
  int wave = threadIdx.x >> 5, lane = threadIdx.x & 31;
  int row = blockIdx.x * 8 + wave;
  if (row >= kNA) return;
  float4 uf[3], wr[3];
  for (int t = 0; t < 3; ++t) {
    uf[t] = ((const float4*)(vsrc + (2 * t) * 128))[lane];
    wr[t] = ((const float4*)(vdst + (2 * t + 1) * 128))[lane];
  }
  float4 xv = ((const float4*)(xA + (long)row * 128))[lane];
  float p[6];
  for (int t = 0; t < 3; ++t) {
    p[t]     = xv.x * uf[t].x + xv.y * uf[t].y + xv.z * uf[t].z + xv.w * uf[t].w;
    p[3 + t] = xv.x * wr[t].x + xv.y * wr[t].y + xv.z * wr[t].z + xv.w * wr[t].w;
  }
  for (int off = 16; off > 0; off >>= 1)
    for (int j = 0; j < 6; ++j) p[j] += __shfl_xor(p[j], off, 32);
  if (lane == 0)
    for (int t = 0; t < 3; ++t) { sFwd[t * kNA + row] = p[t]; sRev[t * kNA + row] = p[3 + t]; }
}

// ------------------------------------------------------------------
// Stage C: edge pass 1 — exp(leaky_relu(logit)) + softmax denominators.
__global__ __launch_bounds__(256) void k_edge1(
    const int* __restrict__ srcA, const int* __restrict__ dstS,
    const float* __restrict__ sFwd_t, const float* __restrict__ dScal_t,
    const float* __restrict__ sRevS_t, const float* __restrict__ sRev_t,
    float* __restrict__ wbF, float* __restrict__ wbR,
    float* __restrict__ denF_t, float* __restrict__ denR_t, int cnt) {
  __shared__ float lden[64];
  if (threadIdx.x < 64) lden[threadIdx.x] = 0.f;
  __syncthreads();
  for (int e = blockIdx.x * blockDim.x + threadIdx.x; e < kE; e += gridDim.x * blockDim.x) {
    int a = srcA[e], s = dstS[e];
    float ef = sFwd_t[a] + dScal_t[s]; ef = ef > 0.f ? ef : 0.2f * ef;
    float wf = __expf(ef); wbF[e] = wf;
    atomicAdd(&lden[s], wf);
    float er = sRevS_t[s] + sRev_t[a]; er = er > 0.f ? er : 0.2f * er;
    float wv = __expf(er); wbR[e] = wv;
    atomicAdd(&denR_t[a], wv);
  }
  __syncthreads();
  if ((int)threadIdx.x < cnt) atomicAdd(&denF_t[threadIdx.x], lden[threadIdx.x]);
}

// ------------------------------------------------------------------
// Stage D: edge pass 2 — alpha-weighted scatter (wave per edge).
__global__ __launch_bounds__(256) void k_edge2(
    const int* __restrict__ srcA, const int* __restrict__ dstS,
    const float* __restrict__ xA, const float* __restrict__ hsRev_t,
    const float* __restrict__ wbF, const float* __restrict__ wbR,
    const float* __restrict__ denF_t, const float* __restrict__ denR_t,
    float* __restrict__ zraw_t, float* __restrict__ zA, int cnt) {
  __shared__ float zacc[kNS * 128];
  for (int i = threadIdx.x; i < cnt * 128; i += blockDim.x) zacc[i] = 0.f;
  __syncthreads();
  int wave = threadIdx.x >> 5, lane = threadIdx.x & 31;
  int nw = gridDim.x * (blockDim.x >> 5);
  for (int e = blockIdx.x * (blockDim.x >> 5) + wave; e < kE; e += nw) {
    int a = srcA[e], s = dstS[e];
    float alf = wbF[e] / fmaxf(denF_t[s], 1e-16f);
    float alr = wbR[e] / fmaxf(denR_t[a], 1e-16f);
    float4 xv = ((const float4*)(xA + (long)a * 128))[lane];
    float4 hv = ((const float4*)(hsRev_t + s * 128))[lane];
    int c0 = lane * 4;
    atomicAdd(&zacc[s * 128 + c0 + 0], alf * xv.x);
    atomicAdd(&zacc[s * 128 + c0 + 1], alf * xv.y);
    atomicAdd(&zacc[s * 128 + c0 + 2], alf * xv.z);
    atomicAdd(&zacc[s * 128 + c0 + 3], alf * xv.w);
    float* za = zA + (long)a * 128 + c0;
    atomicAdd(za + 0, alr * hv.x);
    atomicAdd(za + 1, alr * hv.y);
    atomicAdd(za + 2, alr * hv.z);
    atomicAdd(za + 3, alr * hv.w);
  }
  __syncthreads();
  for (int i = threadIdx.x; i < cnt * 128; i += blockDim.x) atomicAdd(&zraw_t[i], zacc[i]);
}

// ------------------------------------------------------------------
// Stage E1: cv = tanh(zraw @ Wsrc[i] + bias_gat[i])   (59x128 tiny GEMM)
__global__ __launch_bounds__(128) void k_cv(
    const float* __restrict__ zraw, const float* __restrict__ Wsrc,
    const float* __restrict__ bias_gat, float* __restrict__ cv) {
  int n = blockIdx.x, h = threadIdx.x;
  int i = (n < 32) ? 0 : (n < 50 ? 2 : 4);
  const float* Wr = Wsrc + i * 16384;
  float acc = bias_gat[i * 128 + h];
  for (int d = 0; d < 128; ++d) acc += zraw[n * 128 + d] * Wr[d * 128 + h];
  cv[n * 128 + h] = tanhf(acc);
}

// Stage E2: z_a = tanh(acc + b1+b3+b5), in place in d_out.
__global__ void k_zafin(float* __restrict__ zA, const float* __restrict__ bsum) {
  int i = blockIdx.x * blockDim.x + threadIdx.x;
  if (i < kNA * 128) zA[i] = tanhf(zA[i] + bsum[i & 127]);
}

// ------------------------------------------------------------------
// Stage F: biasN = bh + cv @ Wh[128:,:]  (split-K GEMV, reads Wh tail once).
// Explicit global_prefetch to cover the 3840-float row stride.
__global__ __launch_bounds__(256) void k_gemv(
    const float* __restrict__ cv, const float* __restrict__ Wh, float* __restrict__ part) {
  int c = blockIdx.x * 256 + threadIdx.x;
  int r0 = blockIdx.y * 236;
  float acc = 0.f;
  for (int j = 0; j < 236; ++j) {
    int r = r0 + j;
    int rp = r + 12 <= 7551 ? r + 12 : 7551;
    __builtin_prefetch(&Wh[(long)(128 + rp) * kNSH + c], 0, 1);
    acc += cv[r] * Wh[(long)(128 + r) * kNSH + c];
  }
  part[blockIdx.y * kNSH + c] = acc;
}

__global__ __launch_bounds__(256) void k_gemvred(
    const float* __restrict__ part, const float* __restrict__ bh, float* __restrict__ biasN) {
  int c = blockIdx.x * 256 + threadIdx.x;
  float a = bh[c];
  for (int rs = 0; rs < 32; ++rs) a += part[rs * kNSH + c];
  biasN[c] = a;
}

// ------------------------------------------------------------------
// Stage G: fused  heads = relu(x @ Wh[:128] + biasN) @ W_head + b_head.
// WMMA bf16 16x16x32 for both GEMMs; `shared` never touches HBM.
// Wh chunks are staged by the Tensor Data Mover one iteration ahead
// (TENSORcnt pipeline), so the DMA overlaps the WMMA work.
__global__ __launch_bounds__(128) void k_fused_mlp(
    const float* __restrict__ x, const float* __restrict__ Wh,
    const float* __restrict__ biasN,
    const float* __restrict__ Wst, const float* __restrict__ bst,
    const float* __restrict__ Wge, const float* __restrict__ bge,
    const float* __restrict__ Wem, const float* __restrict__ bem,
    float* __restrict__ out) {
  __shared__ __align__(16) __bf16 sX[64][136];   // X tile (rows x K)
  __shared__ __align__(16) __bf16 sB[128][136];  // Wh0 chunk transposed [n][k]
  __shared__ __align__(16) __bf16 sS[64][136];   // relu(shared) chunk [m][n_local]
  __shared__ __align__(16) __bf16 sW[64][136];   // head weights transposed [c][n_local]
#if USE_TDM
  __shared__ __align__(16) float stage[128 * 128]; // fp32 TDM staging (64KB)
#endif

  const int tid = threadIdx.x;
  const int wave = tid >> 5, lane = tid & 31, hf = lane >> 4, lq = lane & 15;
  const int r0 = blockIdx.x * 64;
  const v8f zero8 = {0.f, 0.f, 0.f, 0.f, 0.f, 0.f, 0.f, 0.f};

  for (int idx = tid; idx < 64 * 128; idx += 128) {
    int m = idx >> 7, k = idx & 127;
    sX[m][k] = (__bf16)x[(long)(r0 + m) * kD + k];
  }
#if USE_TDM
  if (tid < 32)   // wave 0 issues the DMA for chunk 0 (EXEC-independent op)
    tdm_load_tile_f32(Wh, (unsigned)(uintptr_t)(void*)stage, 128, 128, kNSH);
#endif
  __syncthreads();

  const int ml = wave * 16 + lq;
  v16bf afrag[4];
  for (int kc = 0; kc < 4; ++kc) {              // A: 16x32, two 8-elem runs per lane
    const __bf16* p = &sX[ml][kc * 32 + hf * 8];
    afrag[kc] = cat16(*(const v8bf*)p, *(const v8bf*)(p + 16));
  }

  v8f hacc[4] = {zero8, zero8, zero8, zero8};   // 64 head cols, persistent over K

  for (int nc = 0; nc < 30; ++nc) {
#if USE_TDM
    if (tid < 32) __builtin_amdgcn_s_wait_tensorcnt(0);  // chunk nc landed in LDS
    __syncthreads();
    for (int idx = tid; idx < 128 * 128; idx += 128) {   // fp32 LDS -> bf16 [n][k]
      int k = idx >> 7, n = idx & 127;
      sB[n][k] = (__bf16)stage[idx];
    }
#else
    for (int idx = tid; idx < 128 * 128; idx += 128) {
      int k = idx >> 7, n = idx & 127;
      sB[n][k] = (__bf16)Wh[(long)k * kNSH + nc * 128 + n];
    }
#endif
    for (int idx = tid; idx < 64 * 128; idx += 128) {
      int c = idx >> 7, kl = idx & 127;
      int kg = nc * 128 + kl;
      float v = 0.f;
      if (c < 32)      v = Wst[kg * 32 + c];
      else if (c < 50) v = Wge[kg * 18 + (c - 32)];
      else if (c < 59) v = Wem[kg * 9 + (c - 50)];
      sW[c][kl] = (__bf16)v;
    }
    __syncthreads();
#if USE_TDM
    if (nc + 1 < 30 && tid < 32)                 // prefetch next chunk via TDM
      tdm_load_tile_f32(Wh + (nc + 1) * 128, (unsigned)(uintptr_t)(void*)stage,
                        128, 128, kNSH);
#endif

    // GEMM1: S = relu(X @ Wh0_chunk + biasN)
    for (int nt = 0; nt < 8; ++nt) {
      v8f acc = zero8;
      int n0 = nt * 16;
      for (int kc = 0; kc < 4; ++kc) {
        const __bf16* p = &sB[n0 + lq][kc * 32 + hf * 8];  // B: two 8-elem runs per lane
        v16bf bfrag = cat16(*(const v8bf*)p, *(const v8bf*)(p + 16));
        acc = __builtin_amdgcn_wmma_f32_16x16x32_bf16(false, afrag[kc], false, bfrag,
                                                      (short)0, acc, false, false);
      }
      float bias = biasN[nc * 128 + n0 + lq];
#pragma unroll
      for (int r = 0; r < 8; ++r) {
        float v = acc[r] + bias;
        v = v > 0.f ? v : 0.f;
        sS[wave * 16 + r + 8 * hf][n0 + lq] = (__bf16)v;
      }
    }
    __syncthreads();

    // GEMM2: hacc += S_chunk @ Whead_chunk
    for (int ct = 0; ct < 4; ++ct) {
      for (int kc = 0; kc < 4; ++kc) {
        const __bf16* pa = &sS[ml][kc * 32 + hf * 8];
        v16bf a2 = cat16(*(const v8bf*)pa, *(const v8bf*)(pa + 16));
        const __bf16* pb = &sW[ct * 16 + lq][kc * 32 + hf * 8];
        v16bf b2 = cat16(*(const v8bf*)pb, *(const v8bf*)(pb + 16));
        hacc[ct] = __builtin_amdgcn_wmma_f32_16x16x32_bf16(false, a2, false, b2,
                                                           (short)0, hacc[ct], false, false);
      }
    }
    __syncthreads();
  }

  for (int ct = 0; ct < 4; ++ct) {
    int c = ct * 16 + lq;
#pragma unroll
    for (int r = 0; r < 8; ++r) {
      int m = r0 + wave * 16 + r + 8 * hf;
      float v = hacc[ct][r];
      if (c < 32)      out[kOutStyle   + m * kNS  + c]        = v + bst[c];
      else if (c < 50) out[kOutGenre   + m * kNG  + (c - 32)] = v + bge[c - 32];
      else if (c < 59) out[kOutEmotion + m * kNEM + (c - 50)] = v + bem[c - 50];
    }
  }
}

// ------------------------------------------------------------------
extern "C" void kernel_launch(void* const* d_in, const int* in_sizes, int n_in,
                              void* d_out, int out_size, void* d_ws, size_t ws_size,
                              hipStream_t stream) {
  const float* x       = (const float*)d_in[0];
  const float* xA      = (const float*)d_in[1];
  const float* xs_s    = (const float*)d_in[2];
  const float* xs_g    = (const float*)d_in[3];
  const float* xs_e    = (const float*)d_in[4];
  const int* es_src    = (const int*)d_in[5];
  const int* es_dst    = (const int*)d_in[6];
  const int* eg_src    = (const int*)d_in[7];
  const int* eg_dst    = (const int*)d_in[8];
  const int* ee_src    = (const int*)d_in[9];
  const int* ee_dst    = (const int*)d_in[10];
  const float* Wsrc    = (const float*)d_in[11];
  const float* Wdst    = (const float*)d_in[12];
  const float* att_src = (const float*)d_in[13];
  const float* att_dst = (const float*)d_in[14];
  const float* bias_gat= (const float*)d_in[15];
  const float* Wh      = (const float*)d_in[16];
  const float* bh      = (const float*)d_in[17];
  const float* Wst     = (const float*)d_in[18];
  const float* bst     = (const float*)d_in[19];
  const float* Wge     = (const float*)d_in[20];
  const float* bge     = (const float*)d_in[21];
  const float* Wem     = (const float*)d_in[22];
  const float* bem     = (const float*)d_in[23];
  float* out = (float*)d_out;
  float* ws  = (float*)d_ws;
  float* zA  = out + kOutZA;

  // zero accumulators (z_a region of d_out, denominators, zraw)
  k_zero<<<(kNA * 128 + 255) / 256, 256, 0, stream>>>(zA, kNA * 128);
  k_zero<<<(300064 + 255) / 256, 256, 0, stream>>>(ws + WS_DENF, 300064);
  k_zero<<<(kCV + 255) / 256, 256, 0, stream>>>(ws + WS_ZRAW, kCV);

  k_small<<<1, 256, 0, stream>>>(xs_s, xs_g, xs_e, Wsrc, Wdst, att_src, att_dst, bias_gat, ws);
  k_rowscal<<<(kNA + 7) / 8, 256, 0, stream>>>(xA, ws + WS_VSRC, ws + WS_VDST,
                                               ws + WS_SFWD, ws + WS_SREV);

  const int* srcs[3]  = {es_src, eg_src, ee_src};
  const int* dsts[3]  = {es_dst, eg_dst, ee_dst};
  const int  cnts[3]  = {kNS, kNG, kNEM};
  const int  doffs[3] = {0, 32, 50};
  for (int t = 0; t < 3; ++t) {
    k_edge1<<<512, 256, 0, stream>>>(srcs[t], dsts[t],
        ws + WS_SFWD + t * kNA, ws + WS_DSCAL + doffs[t],
        ws + WS_SREVS + doffs[t], ws + WS_SREV + t * kNA,
        ws + WS_WBF + t * kE, ws + WS_WBR + t * kE,
        ws + WS_DENF + doffs[t], ws + WS_DENR + t * kNA, cnts[t]);
  }
  for (int t = 0; t < 3; ++t) {
    k_edge2<<<1024, 256, 0, stream>>>(srcs[t], dsts[t], xA,
        ws + WS_HSREV + doffs[t] * 128,
        ws + WS_WBF + t * kE, ws + WS_WBR + t * kE,
        ws + WS_DENF + doffs[t], ws + WS_DENR + t * kNA,
        ws + WS_ZRAW + doffs[t] * 128, zA, cnts[t]);
  }
  k_cv<<<59, 128, 0, stream>>>(ws + WS_ZRAW, Wsrc, bias_gat, ws + WS_CVB);
  k_zafin<<<(kNA * 128 + 255) / 256, 256, 0, stream>>>(zA, ws + WS_BSUM);

  dim3 gGemv(15, 32);
  k_gemv<<<gGemv, 256, 0, stream>>>(ws + WS_CVB, Wh, ws + WS_PART);
  k_gemvred<<<15, 256, 0, stream>>>(ws + WS_PART, bh, ws + WS_BIASN);

  k_fused_mlp<<<kB / 64, 128, 0, stream>>>(x, Wh, ws + WS_BIASN,
                                           Wst, bst, Wge, bge, Wem, bem, out);
}
